// TransformerBlock_644245095085
// MI455X (gfx1250) — matrix operand order repaired and checked by reading, not compiled, not run on hardware
//
#include <hip/hip_runtime.h>

#define B_    2
#define L_    2048
#define DIM_  2048
#define H_    16
#define KVH_  4
#define HD_   128
#define E_    8
#define HID_  8192
#define T_    (B_*L_)
#define MAXROWS_ (T_*2 + E_*64)     /* 8704: assignments padded to 64-row tiles */
#define MAXTILES_ (MAXROWS_/64)     /* 136 */
#define BPAD_ 40                    /* padded LDS row stride (80B, 16B-aligned) */

typedef __bf16 bf16;
typedef __attribute__((ext_vector_type(2)))  __bf16 bf16x2;
typedef __attribute__((ext_vector_type(4)))  __bf16 bf16x4;
typedef __attribute__((ext_vector_type(8)))  __bf16 bf16x8;
typedef __attribute__((ext_vector_type(16))) __bf16 v16bf;
typedef __attribute__((ext_vector_type(8)))  float  v8f;

static __device__ inline v8f wmma_bf16(v16bf a, v16bf b, v8f c) {
  return __builtin_amdgcn_wmma_f32_16x16x32_bf16(false, a, false, b, (short)0, c, false, false);
}

// Build a 16-element fragment from two contiguous 8x bf16 chunks (2 x ds_load_b128)
static __device__ inline v16bf frag16(const bf16* p0, const bf16* p1) {
  bf16x8 lo = *(const bf16x8*)p0;
  bf16x8 hi = *(const bf16x8*)p1;
  v16bf f;
#pragma unroll
  for (int i = 0; i < 8; ++i) { f[i] = lo[i]; f[i + 8] = hi[i]; }
  return f;
}
// A fragment: row-major A row, halves at +koff and +koff+8 (koff = 16 for lanes>=16)
// B fragment: k-contiguous (transposed) row, halves at +kbase and +kbase+8 (kbase = 16 for lanes>=16)

// ---------------------------- prep kernels ----------------------------

__global__ __launch_bounds__(256) void cast_f32_bf16_k(const float* __restrict__ s,
                                                       bf16* __restrict__ d, long n8) {
  long i = (long)blockIdx.x * 256 + threadIdx.x;
  if (i >= n8) return;
  const float4* s4 = (const float4*)s;
  float4 a = s4[2*i], b = s4[2*i + 1];
  bf16x8 o;
  o[0]=(bf16)a.x; o[1]=(bf16)a.y; o[2]=(bf16)a.z; o[3]=(bf16)a.w;
  o[4]=(bf16)b.x; o[5]=(bf16)b.y; o[6]=(bf16)b.z; o[7]=(bf16)b.w;
  ((bf16x8*)d)[i] = o;
}

__global__ __launch_bounds__(256) void pack_qkv_k(const float* __restrict__ wq,
    const float* __restrict__ wk, const float* __restrict__ wv, bf16* __restrict__ dst) {
  long i = (long)blockIdx.x * 256 + threadIdx.x;  // over DIM_*3072
  if (i >= (long)DIM_ * 3072) return;
  int r = (int)(i / 3072), c = (int)(i % 3072);
  float v;
  if (c < 2048)      v = wq[(long)r*2048 + c];
  else if (c < 2560) v = wk[(long)r*512 + (c - 2048)];
  else               v = wv[(long)r*512 + (c - 2560)];
  dst[i] = (bf16)v;
}

__global__ __launch_bounds__(256) void rmsnorm_k(const float* __restrict__ X,
    const float* __restrict__ W, bf16* __restrict__ Y) {
  int t = blockIdx.x;
  const float* x = X + (long)t * DIM_;
  float ss = 0.f;
  for (int i = threadIdx.x; i < DIM_; i += 256) { float v = x[i]; ss += v*v; }
  __shared__ float red[256];
  red[threadIdx.x] = ss;
  __syncthreads();
  for (int s = 128; s > 0; s >>= 1) {
    if (threadIdx.x < s) red[threadIdx.x] += red[threadIdx.x + s];
    __syncthreads();
  }
  float inv = rsqrtf(red[0] / (float)DIM_ + 1e-6f);
  for (int i = threadIdx.x; i < DIM_; i += 256)
    Y[(long)t*DIM_ + i] = (bf16)(x[i] * inv * W[i]);
}

static __device__ inline float rope_val(const float* hp, int d, int l) {
  int i = d & 63;
  float inv = __expf(-(float)i * (9.210340371976184f / 64.0f)); // 10000^(-i/64)
  float ang = (float)l * inv;
  float c = cosf(ang), s = sinf(ang);
  float x = hp[d];
  float other = (d < 64) ? -hp[d + 64] : hp[d - 64];
  return x * c + other * s;
}

__global__ __launch_bounds__(256) void rope_arrange_k(const float* __restrict__ qkv,
    bf16* __restrict__ Qo, bf16* __restrict__ Ko, bf16* __restrict__ Vo) {
  long idx = (long)blockIdx.x * 256 + threadIdx.x;  // T_*3072
  if (idx >= (long)T_ * 3072) return;
  int t = (int)(idx / 3072), c = (int)(idx % 3072);
  int b = t / L_, l = t % L_;
  const float* row = qkv + (long)t * 3072;
  if (c < 2048) {
    int h = c >> 7, d = c & 127;
    Qo[((long)(b*H_ + h)*L_ + l)*HD_ + d] = (bf16)rope_val(row + h*HD_, d, l);
  } else if (c < 2560) {
    int cc = c - 2048; int h = cc >> 7, d = cc & 127;
    Ko[((long)(b*KVH_ + h)*L_ + l)*HD_ + d] = (bf16)rope_val(row + 2048 + h*HD_, d, l);
  } else {
    int cc = c - 2560; int h = cc >> 7, d = cc & 127;
    Vo[((long)(b*KVH_ + h)*L_ + l)*HD_ + d] = (bf16)row[2560 + cc];
  }
}

// V transpose: [bh, l, d] -> [bh, d, l] so attention V-fragments are k-contiguous.
__global__ __launch_bounds__(256) void vtrans_k(const bf16* __restrict__ Vi,
                                                bf16* __restrict__ Vt) {
  int bh = blockIdx.x;
  int l0 = blockIdx.y * 64, d0 = blockIdx.z * 64;
  __shared__ bf16 tile[64][72];
  int tid = threadIdx.x;
  int rr = tid >> 2, c16 = (tid & 3) * 16;
  const bf16* src = Vi + ((long)bh * L_ + l0) * HD_ + d0;
#pragma unroll
  for (int j = 0; j < 2; ++j)
    *(bf16x8*)&tile[rr][c16 + j*8] = *(const bf16x8*)(src + (long)rr*HD_ + c16 + j*8);
  __syncthreads();
  bf16* dst = Vt + ((long)bh * HD_ + d0) * L_ + l0;
#pragma unroll
  for (int j = 0; j < 2; ++j) {
    bf16x8 o;
#pragma unroll
    for (int i = 0; i < 8; ++i) o[i] = tile[c16 + j*8 + i][rr];
    *(bf16x8*)(dst + (long)rr*L_ + c16 + j*8) = o;
  }
}

// ---------------------------- dense GEMM ----------------------------
// C[M,N] f32 (optional +R) = A[M,K]bf16 @ B[K,N]bf16. 64x64x32 tiles, 8 waves,
// B stored k-contiguous (transposed) in LDS, register double buffering.
template<bool RES>
__global__ __launch_bounds__(256) void gemm_k(const bf16* __restrict__ A,
    const bf16* __restrict__ Bm, const float* __restrict__ R, float* __restrict__ C,
    int M, int N, int K) {
  __shared__ bf16 As[64][BPAD_];
  __shared__ bf16 Bt[64][BPAD_];
  const int tid = threadIdx.x;
  const int m0 = blockIdx.x * 64, n0 = blockIdx.y * 64;
  const int lane = tid & 31, wid = tid >> 5;
  const int wm = wid & 3, wn = wid >> 2;
  const int ar = tid >> 2, ac = (tid & 3) * 8;
  const int bk2 = (tid >> 4) * 2, bn4 = (tid & 15) * 4;
  const int arow = wm*16 + (lane & 15);
  const int koff = (lane >> 4) << 4;
  const int kbase = (lane >> 4) << 4;
  const int bcol = wn*32 + (lane & 15);
  v8f acc0 = {0,0,0,0,0,0,0,0}, acc1 = {0,0,0,0,0,0,0,0};
  const bf16* aptr = A + (long)(m0 + ar) * K + ac;
  const bf16* bptr = Bm + (long)bk2 * N + n0 + bn4;
  bf16x8 areg = *(const bf16x8*)aptr;
  bf16x4 breg0 = *(const bf16x4*)bptr;
  bf16x4 breg1 = *(const bf16x4*)(bptr + N);
  for (int k0 = 0; k0 < K; k0 += 32) {
    *(bf16x8*)&As[ar][ac] = areg;
#pragma unroll
    for (int j = 0; j < 4; ++j) {
      bf16x2 pr; pr[0] = breg0[j]; pr[1] = breg1[j];
      *(bf16x2*)&Bt[bn4 + j][bk2] = pr;
    }
    __syncthreads();
    aptr += 32; bptr += (long)32 * N;
    if (k0 + 32 < K) {
      areg  = *(const bf16x8*)aptr;
      breg0 = *(const bf16x4*)bptr;
      breg1 = *(const bf16x4*)(bptr + N);
    }
    v16bf af = frag16(&As[arow][koff], &As[arow][koff + 8]);
    v16bf b0 = frag16(&Bt[bcol][kbase], &Bt[bcol][kbase + 8]);
    v16bf b1 = frag16(&Bt[bcol + 16][kbase], &Bt[bcol + 16][kbase + 8]);
    acc0 = wmma_bf16(af, b0, acc0);
    acc1 = wmma_bf16(af, b1, acc1);
    __syncthreads();
  }
  const int hi8 = (lane >> 4) << 3, cn = lane & 15;
#pragma unroll
  for (int v = 0; v < 8; ++v) {
    int r = m0 + wm*16 + v + hi8;
    int c0 = n0 + wn*32 + cn;
    float o0 = acc0[v], o1 = acc1[v];
    if (RES) { o0 += R[(long)r*N + c0]; o1 += R[(long)r*N + c0 + 16]; }
    C[(long)r*N + c0]      = o0;
    C[(long)r*N + c0 + 16] = o1;
  }
}

// ---------------------------- flash attention ----------------------------
// grid (B_*H_, L_/64). 64 q-rows per WG, 64-key tiles, online softmax.
__global__ __launch_bounds__(256) void attn_k(const bf16* __restrict__ Qb,
    const bf16* __restrict__ Kb, const bf16* __restrict__ VbT, bf16* __restrict__ Out) {
  const int bh = blockIdx.x;
  const int b = bh / H_, h = bh % H_;
  const int kvh = h / (H_ / KVH_);
  const int qt = blockIdx.y, q0 = qt * 64;
  const int tid = threadIdx.x;
  const int lane = tid & 31, wid = tid >> 5;
  const int wm = wid & 3, wn = wid >> 2;        // wn in {0,1}
  const int hi8 = (lane >> 4) << 3, cn = lane & 15;
  const int koff = (lane >> 4) << 4;
  const int kbase = (lane >> 4) << 4;
  const float scale = 0.088388347648318447f;    // 1/sqrt(128)

  __shared__ bf16 Qs[64][HD_];
  __shared__ bf16 KVbuf[64 * HD_];              // K tile [key][128] then V tile [d][64]
  __shared__ float Ss[64][64];
  __shared__ bf16 Ps[64][64];
  __shared__ float mrow[64], lrow[64], arow[64];

  {
    const bf16* src = Qb + ((long)(b*H_ + h)*L_ + q0) * HD_;
#pragma unroll
    for (int j = 0; j < 4; ++j) {
      int vid = tid + j*256; int r = vid >> 4, c = (vid & 15) * 8;
      *(bf16x8*)&Qs[r][c] = *(const bf16x8*)(src + (long)r*HD_ + c);
    }
  }
  if (tid < 64) { mrow[tid] = -__builtin_inff(); lrow[tid] = 0.f; }
  const v8f z8 = {0,0,0,0,0,0,0,0};
  v8f oacc[4];
#pragma unroll
  for (int s = 0; s < 4; ++s) oacc[s] = z8;

  const bf16* ksrc  = Kb  + (long)(b*KVH_ + kvh) * L_ * HD_;
  const bf16* vsrcT = VbT + (long)(b*KVH_ + kvh) * HD_ * L_;

  for (int kt = 0; kt <= qt; ++kt) {
    const int k0 = kt * 64;
#pragma unroll
    for (int j = 0; j < 4; ++j) {               // K tile -> [key][128]
      int vid = tid + j*256; int r = vid >> 4, c = (vid & 15) * 8;
      *(bf16x8*)(KVbuf + r*HD_ + c) = *(const bf16x8*)(ksrc + (long)(k0 + r)*HD_ + c);
    }
    __syncthreads();

    // S = Q @ K^T (k-dim = d = 128 -> 4 WMMA steps); K-tile fragments are d-contiguous
    v8f s0 = z8, s1 = z8;
#pragma unroll
    for (int ks = 0; ks < 4; ++ks) {
      int kb = ks * 32;
      const int qrow = wm*16 + cn;
      v16bf aq = frag16(&Qs[qrow][kb + koff], &Qs[qrow][kb + koff + 8]);
      const bf16* kr0 = KVbuf + (wn*32 + cn) * HD_ + kb + kbase;
      const bf16* kr1 = KVbuf + (wn*32 + 16 + cn) * HD_ + kb + kbase;
      v16bf b0 = frag16(kr0, kr0 + 8);
      v16bf b1 = frag16(kr1, kr1 + 8);
      s0 = wmma_bf16(aq, b0, s0);
      s1 = wmma_bf16(aq, b1, s1);
    }
#pragma unroll
    for (int v = 0; v < 8; ++v) {               // scale + causal mask
      int r = wm*16 + v + hi8;
      int c0 = wn*32 + cn, c1 = c0 + 16;
      float e0 = s0[v] * scale, e1 = s1[v] * scale;
      if (k0 + c0 > q0 + r) e0 = -__builtin_inff();
      if (k0 + c1 > q0 + r) e1 = -__builtin_inff();
      Ss[r][c0] = e0; Ss[r][c1] = e1;
    }
    __syncthreads();

#pragma unroll
    for (int j = 0; j < 4; ++j) {               // V tile -> [d][64] (pre-transposed src)
      int d = tid >> 1, half = tid & 1;
      int c = half*32 + j*8;
      *(bf16x8*)(KVbuf + d*64 + c) = *(const bf16x8*)(vsrcT + (long)d*L_ + k0 + c);
    }
    if (tid < 64) {                             // online softmax (vectorized), P -> bf16
      int r = tid;
      const float4* srow = (const float4*)&Ss[r][0];
      float m_old = mrow[r], tm = m_old;
#pragma unroll
      for (int j = 0; j < 16; ++j) {
        float4 v = srow[j];
        tm = fmaxf(tm, fmaxf(fmaxf(v.x, v.y), fmaxf(v.z, v.w)));
      }
      float alpha = __expf(m_old - tm);
      float sum = 0.f;
#pragma unroll
      for (int j = 0; j < 16; ++j) {
        float4 v = srow[j];
        float p0 = __expf(v.x - tm), p1 = __expf(v.y - tm);
        float p2 = __expf(v.z - tm), p3 = __expf(v.w - tm);
        sum += (p0 + p1) + (p2 + p3);
        bf16x4 pb; pb[0]=(bf16)p0; pb[1]=(bf16)p1; pb[2]=(bf16)p2; pb[3]=(bf16)p3;
        *(bf16x4*)&Ps[r][j*4] = pb;
      }
      lrow[r] = lrow[r] * alpha + sum;
      mrow[r] = tm; arow[r] = alpha;
    }
    __syncthreads();

#pragma unroll
    for (int v = 0; v < 8; ++v) {               // rescale running O
      float al = arow[wm*16 + v + hi8];
#pragma unroll
      for (int s = 0; s < 4; ++s) oacc[s][v] *= al;
    }
#pragma unroll
    for (int ks = 0; ks < 2; ++ks) {            // O += P @ V (keys = 64)
      int kb = ks * 32;
      const int prow = wm*16 + cn;
      v16bf ap = frag16(&Ps[prow][kb + koff], &Ps[prow][kb + koff + 8]);
#pragma unroll
      for (int s = 0; s < 4; ++s) {
        int d = wn*64 + s*16 + cn;
        const bf16* vr = KVbuf + d*64 + kb + kbase;
        v16bf bv = frag16(vr, vr + 8);
        oacc[s] = wmma_bf16(ap, bv, oacc[s]);
      }
    }
    __syncthreads();
  }

#pragma unroll
  for (int v = 0; v < 8; ++v) {                 // O /= l, write [T, H*HD] bf16
    int r = wm*16 + v + hi8;
    float linv = 1.0f / lrow[r];
#pragma unroll
    for (int s = 0; s < 4; ++s) {
      int d = wn*64 + s*16 + cn;
      Out[((long)(b*L_ + q0 + r))*DIM_ + h*HD_ + d] = (bf16)(oacc[s][v] * linv);
    }
  }
}

// ---------------------------- MoE routing ----------------------------

__global__ __launch_bounds__(256) void router_k(const bf16* __restrict__ Xn,
    const float* __restrict__ RW, int* __restrict__ topi, float* __restrict__ topw,
    int* __restrict__ counts) {
  int t = blockIdx.x;
  float acc[E_];
#pragma unroll
  for (int e = 0; e < E_; ++e) acc[e] = 0.f;
  for (int i = threadIdx.x; i < DIM_; i += 256) {
    float xv = (float)Xn[(long)t*DIM_ + i];
#pragma unroll
    for (int e = 0; e < E_; ++e) acc[e] += xv * RW[i*E_ + e];
  }
  __shared__ float red[256][E_];
#pragma unroll
  for (int e = 0; e < E_; ++e) red[threadIdx.x][e] = acc[e];
  __syncthreads();
  for (int s = 128; s > 0; s >>= 1) {
    if (threadIdx.x < s)
      for (int e = 0; e < E_; ++e) red[threadIdx.x][e] += red[threadIdx.x + s][e];
    __syncthreads();
  }
  if (threadIdx.x == 0) {
    float mx = red[0][0];
    for (int e = 1; e < E_; ++e) mx = fmaxf(mx, red[0][e]);
    float pr[E_]; float sum = 0.f;
    for (int e = 0; e < E_; ++e) { pr[e] = __expf(red[0][e] - mx); sum += pr[e]; }
    for (int e = 0; e < E_; ++e) pr[e] /= sum;
    int i0 = 0;
    for (int e = 1; e < E_; ++e) if (pr[e] > pr[i0]) i0 = e;
    int i1 = (i0 == 0) ? 1 : 0;
    for (int e = 0; e < E_; ++e) if (e != i0 && pr[e] > pr[i1]) i1 = e;
    float sw = pr[i0] + pr[i1];
    topi[t*2] = i0; topi[t*2 + 1] = i1;
    topw[t*2] = pr[i0] / sw; topw[t*2 + 1] = pr[i1] / sw;
    atomicAdd(&counts[i0], 1);
    atomicAdd(&counts[i1], 1);
  }
}

__global__ void moe_init_k(int* counts, int* fill, int* rowmap, int* texp) {
  int i = blockIdx.x * 256 + threadIdx.x;
  if (i < E_) { counts[i] = 0; fill[i] = 0; }
  if (i < 256) texp[i] = -1;
  if (i < MAXROWS_) rowmap[i] = -1;
}

__global__ void moe_scan_k(const int* counts, int* offsets, int* texp) {
  if (threadIdx.x == 0 && blockIdx.x == 0) {
    int off = 0;
    for (int e = 0; e < E_; ++e) {
      offsets[e] = off;
      int tl = (counts[e] + 63) >> 6;
      for (int i = 0; i < tl; ++i) texp[(off >> 6) + i] = e;
      off += tl << 6;
    }
  }
}

__global__ void moe_scatter_k(const int* topi, const int* offsets, int* fill,
                              int* rowmap, int* slotpos) {
  int t = blockIdx.x * 256 + threadIdx.x;
  if (t >= T_) return;
#pragma unroll
  for (int s = 0; s < 2; ++s) {
    int e = topi[t*2 + s];
    int p = offsets[e] + atomicAdd(&fill[e], 1);
    rowmap[p] = t;
    slotpos[t*2 + s] = p;
  }
}

// ---------------------------- grouped MoE GEMMs ----------------------------
// GEMM1: Hb[row, HID] = silu(gather(Xn)[row] @ w1[expert(row)])  (bf16 out)
__global__ __launch_bounds__(256) void moe_gemm1_k(const bf16* __restrict__ Xn,
    const bf16* __restrict__ W1, const int* __restrict__ rowmap,
    const int* __restrict__ texp, bf16* __restrict__ Hb) {
  const int mt = blockIdx.x;
  const int e = texp[mt];
  if (e < 0) return;
  const int n0 = blockIdx.y * 64;
  __shared__ bf16 As[64][BPAD_];
  __shared__ bf16 Bt[64][BPAD_];
  const int tid = threadIdx.x;
  const int lane = tid & 31, wid = tid >> 5;
  const int wm = wid & 3, wn = wid >> 2;
  const int ar = tid >> 2, ac = (tid & 3) * 8;
  const int bk2 = (tid >> 4) * 2, bn4 = (tid & 15) * 4;
  const int arow = wm*16 + (lane & 15);
  const int koff = (lane >> 4) << 4;
  const int kbase = (lane >> 4) << 4;
  const int bcol = wn*32 + (lane & 15);
  const int tok = rowmap[mt*64 + ar];
  const bf16* aptr = (tok >= 0) ? (Xn + (long)tok*DIM_ + ac) : nullptr;
  const bf16* bptr = W1 + (long)e*DIM_*HID_ + (long)bk2*HID_ + n0 + bn4;
  v8f acc0 = {0,0,0,0,0,0,0,0}, acc1 = {0,0,0,0,0,0,0,0};
  bf16x8 areg;
#pragma unroll
  for (int i = 0; i < 8; ++i) areg[i] = (bf16)0.f;
  if (aptr) areg = *(const bf16x8*)aptr;
  bf16x4 breg0 = *(const bf16x4*)bptr;
  bf16x4 breg1 = *(const bf16x4*)(bptr + HID_);
  for (int k0 = 0; k0 < DIM_; k0 += 32) {
    *(bf16x8*)&As[ar][ac] = areg;
#pragma unroll
    for (int j = 0; j < 4; ++j) {
      bf16x2 pr; pr[0] = breg0[j]; pr[1] = breg1[j];
      *(bf16x2*)&Bt[bn4 + j][bk2] = pr;
    }
    __syncthreads();
    bptr += (long)32 * HID_;
    if (k0 + 32 < DIM_) {
      if (aptr) { aptr += 32; areg = *(const bf16x8*)aptr; }
      breg0 = *(const bf16x4*)bptr;
      breg1 = *(const bf16x4*)(bptr + HID_);
    }
    v16bf af = frag16(&As[arow][koff], &As[arow][koff + 8]);
    v16bf b0 = frag16(&Bt[bcol][kbase], &Bt[bcol][kbase + 8]);
    v16bf b1 = frag16(&Bt[bcol + 16][kbase], &Bt[bcol + 16][kbase + 8]);
    acc0 = wmma_bf16(af, b0, acc0);
    acc1 = wmma_bf16(af, b1, acc1);
    __syncthreads();
  }
  const int hi8 = (lane >> 4) << 3, cn = lane & 15;
#pragma unroll
  for (int v = 0; v < 8; ++v) {
    int r = mt*64 + wm*16 + v + hi8;
    int c0 = n0 + wn*32 + cn;
    float x0 = acc0[v], x1 = acc1[v];
    x0 = x0 / (1.f + __expf(-x0));              // silu
    x1 = x1 / (1.f + __expf(-x1));
    Hb[(long)r*HID_ + c0]      = (bf16)x0;
    Hb[(long)r*HID_ + c0 + 16] = (bf16)x1;
  }
}

// GEMM2: Mo[row, DIM] = Hb[row] @ w2[expert(row)]  (f32 out)
__global__ __launch_bounds__(256) void moe_gemm2_k(const bf16* __restrict__ Hb,
    const bf16* __restrict__ W2, const int* __restrict__ texp, float* __restrict__ Mo) {
  const int mt = blockIdx.x;
  const int e = texp[mt];
  if (e < 0) return;
  const int n0 = blockIdx.y * 64;
  __shared__ bf16 As[64][BPAD_];
  __shared__ bf16 Bt[64][BPAD_];
  const int tid = threadIdx.x;
  const int lane = tid & 31, wid = tid >> 5;
  const int wm = wid & 3, wn = wid >> 2;
  const int ar = tid >> 2, ac = (tid & 3) * 8;
  const int bk2 = (tid >> 4) * 2, bn4 = (tid & 15) * 4;
  const int arow = wm*16 + (lane & 15);
  const int koff = (lane >> 4) << 4;
  const int kbase = (lane >> 4) << 4;
  const int bcol = wn*32 + (lane & 15);
  const bf16* aptr = Hb + (long)(mt*64 + ar)*HID_ + ac;
  const bf16* bptr = W2 + (long)e*HID_*DIM_ + (long)bk2*DIM_ + n0 + bn4;
  v8f acc0 = {0,0,0,0,0,0,0,0}, acc1 = {0,0,0,0,0,0,0,0};
  bf16x8 areg = *(const bf16x8*)aptr;
  bf16x4 breg0 = *(const bf16x4*)bptr;
  bf16x4 breg1 = *(const bf16x4*)(bptr + DIM_);
  for (int k0 = 0; k0 < HID_; k0 += 32) {
    *(bf16x8*)&As[ar][ac] = areg;
#pragma unroll
    for (int j = 0; j < 4; ++j) {
      bf16x2 pr; pr[0] = breg0[j]; pr[1] = breg1[j];
      *(bf16x2*)&Bt[bn4 + j][bk2] = pr;
    }
    __syncthreads();
    aptr += 32; bptr += (long)32 * DIM_;
    if (k0 + 32 < HID_) {
      areg  = *(const bf16x8*)aptr;
      breg0 = *(const bf16x4*)bptr;
      breg1 = *(const bf16x4*)(bptr + DIM_);
    }
    v16bf af = frag16(&As[arow][koff], &As[arow][koff + 8]);
    v16bf b0 = frag16(&Bt[bcol][kbase], &Bt[bcol][kbase + 8]);
    v16bf b1 = frag16(&Bt[bcol + 16][kbase], &Bt[bcol + 16][kbase + 8]);
    acc0 = wmma_bf16(af, b0, acc0);
    acc1 = wmma_bf16(af, b1, acc1);
    __syncthreads();
  }
  const int hi8 = (lane >> 4) << 3, cn = lane & 15;
#pragma unroll
  for (int v = 0; v < 8; ++v) {
    int r = mt*64 + wm*16 + v + hi8;
    int c0 = n0 + wn*32 + cn;
    Mo[(long)r*DIM_ + c0]      = acc0[v];
    Mo[(long)r*DIM_ + c0 + 16] = acc1[v];
  }
}

__global__ __launch_bounds__(256) void combine_k(const float* __restrict__ Hres,
    const float* __restrict__ Mo, const int* __restrict__ slotpos,
    const float* __restrict__ topw, float* __restrict__ Out) {
  long idx = (long)blockIdx.x * 256 + threadIdx.x;
  if (idx >= (long)T_ * DIM_) return;
  int t = (int)(idx / DIM_), d = (int)(idx % DIM_);
  int p0 = slotpos[t*2], p1 = slotpos[t*2 + 1];
  Out[idx] = Hres[idx] + topw[t*2]     * Mo[(long)p0*DIM_ + d]
                       + topw[t*2 + 1] * Mo[(long)p1*DIM_ + d];
}

// ---------------------------- launch ----------------------------

extern "C" void kernel_launch(void* const* d_in, const int* in_sizes, int n_in,
                              void* d_out, int out_size, void* d_ws, size_t ws_size,
                              hipStream_t stream) {
  (void)in_sizes; (void)n_in; (void)out_size; (void)ws_size;
  const float* x   = (const float*)d_in[0];
  const float* n1w = (const float*)d_in[1];
  const float* wq  = (const float*)d_in[2];
  const float* wk  = (const float*)d_in[3];
  const float* wv  = (const float*)d_in[4];
  const float* wo  = (const float*)d_in[5];
  const float* n2w = (const float*)d_in[6];
  const float* rw  = (const float*)d_in[7];
  const float* w1  = (const float*)d_in[8];
  const float* w2  = (const float*)d_in[9];
  float* out = (float*)d_out;

  char* p = (char*)d_ws;
  auto alloc = [&](size_t bytes) -> void* {
    void* r = (void*)p;
    p += (bytes + 255) & ~(size_t)255;
    return r;
  };
  bf16*  wqkv_b = (bf16*) alloc((size_t)DIM_*3072*2);
  bf16*  wo_b   = (bf16*) alloc((size_t)DIM_*DIM_*2);
  bf16*  w1_b   = (bf16*) alloc((size_t)E_*DIM_*HID_*2);
  bf16*  w2_b   = (bf16*) alloc((size_t)E_*HID_*DIM_*2);
  bf16*  xn_b   = (bf16*) alloc((size_t)T_*DIM_*2);
  float* qkv_f  = (float*)alloc((size_t)T_*3072*4);
  bf16*  q_b    = (bf16*) alloc((size_t)T_*DIM_*2);
  bf16*  k_b    = (bf16*) alloc((size_t)B_*KVH_*L_*HD_*2);
  bf16*  v_b    = (bf16*) alloc((size_t)B_*KVH_*L_*HD_*2);
  bf16*  vT_b   = (bf16*) alloc((size_t)B_*KVH_*HD_*L_*2);
  bf16*  ao_b   = (bf16*) alloc((size_t)T_*DIM_*2);
  float* h_f    = (float*)alloc((size_t)T_*DIM_*4);
  bf16*  hn_b   = (bf16*) alloc((size_t)T_*DIM_*2);
  int*   topi   = (int*)  alloc((size_t)T_*2*4);
  float* topw   = (float*)alloc((size_t)T_*2*4);
  int*   counts = (int*)  alloc(E_*4);
  int*   fill   = (int*)  alloc(E_*4);
  int*   offs   = (int*)  alloc(E_*4);
  int*   texp   = (int*)  alloc(256*4);
  int*   rowmap = (int*)  alloc((size_t)MAXROWS_*4);
  int*   slotp  = (int*)  alloc((size_t)T_*2*4);
  bf16*  hbuf   = (bf16*) alloc((size_t)MAXROWS_*HID_*2);
  float* moeo   = (float*)alloc((size_t)MAXROWS_*DIM_*4);

  // weight precision conversion (per-launch, deterministic)
  pack_qkv_k<<<(int)(((long)DIM_*3072 + 255)/256), 256, 0, stream>>>(wq, wk, wv, wqkv_b);
  cast_f32_bf16_k<<<(int)(((long)DIM_*DIM_/8 + 255)/256), 256, 0, stream>>>(wo, wo_b, (long)DIM_*DIM_/8);
  cast_f32_bf16_k<<<(int)(((long)E_*DIM_*HID_/8 + 255)/256), 256, 0, stream>>>(w1, w1_b, (long)E_*DIM_*HID_/8);
  cast_f32_bf16_k<<<(int)(((long)E_*HID_*DIM_/8 + 255)/256), 256, 0, stream>>>(w2, w2_b, (long)E_*HID_*DIM_/8);
  moe_init_k<<<(MAXROWS_ + 255)/256, 256, 0, stream>>>(counts, fill, rowmap, texp);

  // attention path
  rmsnorm_k<<<T_, 256, 0, stream>>>(x, n1w, xn_b);
  gemm_k<false><<<dim3(T_/64, 3072/64), 256, 0, stream>>>(xn_b, wqkv_b, nullptr, qkv_f, T_, 3072, DIM_);
  rope_arrange_k<<<(int)(((long)T_*3072 + 255)/256), 256, 0, stream>>>(qkv_f, q_b, k_b, v_b);
  vtrans_k<<<dim3(B_*KVH_, L_/64, HD_/64), 256, 0, stream>>>(v_b, vT_b);
  attn_k<<<dim3(B_*H_, L_/64), 256, 0, stream>>>(q_b, k_b, vT_b, ao_b);
  gemm_k<true><<<dim3(T_/64, DIM_/64), 256, 0, stream>>>(ao_b, wo_b, x, h_f, T_, DIM_, DIM_);

  // MoE path
  rmsnorm_k<<<T_, 256, 0, stream>>>(h_f, n2w, hn_b);
  router_k<<<T_, 256, 0, stream>>>(hn_b, rw, topi, topw, counts);
  moe_scan_k<<<1, 32, 0, stream>>>(counts, offs, texp);
  moe_scatter_k<<<(T_ + 255)/256, 256, 0, stream>>>(topi, offs, fill, rowmap, slotp);
  moe_gemm1_k<<<dim3(MAXTILES_, HID_/64), 256, 0, stream>>>(hn_b, w1_b, rowmap, texp, hbuf);
  moe_gemm2_k<<<dim3(MAXTILES_, DIM_/64), 256, 0, stream>>>(hbuf, w2_b, texp, moeo);
  combine_k<<<(int)(((long)T_*DIM_ + 255)/256), 256, 0, stream>>>(h_f, moeo, slotp, topw, out);
}
